// UpBlock_14937896256223
// MI455X (gfx1250) — compile-verified, matrix operand order checked by reading, not run on hardware
//
#include <hip/hip_runtime.h>

typedef __bf16 bf16_t;
typedef __attribute__((ext_vector_type(16))) __bf16 v16bf;
typedef __attribute__((ext_vector_type(8)))  float  v8f;

#define C_IN  128
#define C_OUT 64
#define BN_EPS 1e-5f

union Frag16 { v16bf v; uint4 q[2]; };

__device__ __forceinline__ v8f wmma_bf16(const Frag16& a, const Frag16& b, v8f c) {
  return __builtin_amdgcn_wmma_f32_16x16x32_bf16(false, a.v, false, b.v,
                                                 (short)0, c, false, false);
}

// K-index inside a 32-wide K step for element i of a lane's fragment.
// Per ISA 16-bit A/B layout: lane-half h holds K = 8h..8h+7 (elems 0..7)
// and K = 16+8h..23+8h (elems 8..15).
__device__ __forceinline__ int kin_of(int i, int h) {
  return (i < 8) ? (8 * h + i) : (16 + 8 * h + (i - 8));
}

// ---------------- weight pre-swizzle into per-lane WMMA B layout ------------
// Bup pack layout:   [ntileG(32)][kchunk(4)][lane(32)][16]  (B = W_up as [128 x 512])
__global__ void pack_wup(const float* __restrict__ Wup, bf16_t* __restrict__ dst) {
  int g = blockIdx.x * blockDim.x + threadIdx.x;     // 32*4*32 = 4096 threads
  int lane = g & 31;
  int kchunk = (g >> 5) & 3;
  int ntileG = g >> 7;
  int h = lane >> 4;
  int col = ntileG * 16 + (lane & 15);               // 0..511
  int oct = col >> 6, d = col & 63;
  bf16_t* o = dst + (size_t)g * 16;
#pragma unroll
  for (int i = 0; i < 16; ++i) {
    int K = kchunk * 32 + kin_of(i, h);
    o[i] = (bf16_t)Wup[((size_t)oct * C_IN + K) * C_OUT + d];
  }
}

// Bconv pack layout: [tap(27)][ntile(4)][kchunk(2)][lane(32)][16]
__global__ void pack_wconv(const float* __restrict__ Wconv, bf16_t* __restrict__ dst) {
  int g = blockIdx.x * blockDim.x + threadIdx.x;     // 27*4*2*32 = 6912 threads
  int lane = g & 31;
  int kc = (g >> 5) & 1;
  int nt = (g >> 6) & 3;
  int tap = g >> 8;
  int h = lane >> 4;
  int col = nt * 16 + (lane & 15);                   // 0..63
  bf16_t* o = dst + (size_t)g * 16;
#pragma unroll
  for (int i = 0; i < 16; ++i) {
    int K = kc * 32 + kin_of(i, h);
    o[i] = (bf16_t)Wconv[((size_t)tap * C_OUT + K) * C_OUT + col];
  }
}

// ---------------- deconv GEMM:  Y[65536,512] = X[65536,128] @ Bup ----------
// Each wave: 2 M-tiles (32 parents) x 64 N; B fragments amortized over 2 WMMAs.
__global__ void __launch_bounds__(256) gemm_up(const float* __restrict__ x,
                                               const bf16_t* __restrict__ Bp,
                                               bf16_t* __restrict__ Ybf) {
  int lane = threadIdx.x & 31;
  int w = threadIdx.x >> 5;                          // 0..7 = N block
  int l15 = lane & 15, h = lane >> 4;
  size_t base = (size_t)blockIdx.x * 32;             // parent base (2 tiles)
  const float* xrow0 = x + (base + l15) * C_IN;
  const float* xrow1 = x + (base + 16 + l15) * C_IN;

  v8f acc[8];                                        // [mt*4 + nt]
#pragma unroll
  for (int a = 0; a < 8; ++a)
#pragma unroll
    for (int j = 0; j < 8; ++j) acc[a][j] = 0.f;

#pragma unroll
  for (int kk = 0; kk < 4; ++kk) {                   // K = kk*32
    Frag16 a0, a1;
    const float* p0 = xrow0 + kk * 32 + 8 * h;       // K = kk*32 + 8h .. +7
    const float* p1 = xrow1 + kk * 32 + 8 * h;
#pragma unroll
    for (int i = 0; i < 8; ++i) {
      a0.v[i] = (bf16_t)p0[i];      a0.v[8 + i] = (bf16_t)p0[16 + i];
      a1.v[i] = (bf16_t)p1[i];      a1.v[8 + i] = (bf16_t)p1[16 + i];
    }
#pragma unroll
    for (int nt = 0; nt < 4; ++nt) {
      int ntg = w * 4 + nt;
      const uint4* bq = (const uint4*)(Bp + ((size_t)(ntg * 4 + kk) * 32 + lane) * 16);
      Frag16 b; b.q[0] = bq[0]; b.q[1] = bq[1];
      acc[nt]     = wmma_bf16(a0, b, acc[nt]);
      acc[4 + nt] = wmma_bf16(a1, b, acc[4 + nt]);
    }
  }
  // C layout: VGPR j -> (M = j + 8h, N = lane%16)
#pragma unroll
  for (int mt = 0; mt < 2; ++mt)
#pragma unroll
    for (int nt = 0; nt < 4; ++nt)
#pragma unroll
      for (int j = 0; j < 8; ++j) {
        size_t parent = base + mt * 16 + j + 8 * h;
        int col = w * 64 + nt * 16 + l15;
        Ybf[parent * 512 + col] = (bf16_t)acc[mt * 4 + nt][j];
      }
}

// ---------------- per-channel batch stats (sum, sumsq) ---------------------
template <typename T>
__global__ void __launch_bounds__(256) bn_stats(const T* __restrict__ src, long long nrows,
                                                float* __restrict__ gsum,
                                                float* __restrict__ gsq) {
  __shared__ float ssum[C_OUT], ssq[C_OUT];
  int tid = threadIdx.x;
  if (tid < C_OUT) { ssum[tid] = 0.f; ssq[tid] = 0.f; }
  __syncthreads();
  long long g = (long long)blockIdx.x * blockDim.x + tid;
  long long T64 = ((long long)gridDim.x * blockDim.x) >> 6;
  int c = (int)(g & 63);
  float s = 0.f, q = 0.f;
  for (long long r = g >> 6; r < nrows; r += T64) {
    float v = (float)src[r * 64 + c];
    s += v; q += v * v;
  }
  atomicAdd(&ssum[c], s);
  atomicAdd(&ssq[c], q);
  __syncthreads();
  if (tid < C_OUT) { atomicAdd(&gsum[tid], ssum[tid]); atomicAdd(&gsq[tid], ssq[tid]); }
}

// ---------------- BN + ELU applied in place --------------------------------
template <typename T>
__global__ void __launch_bounds__(256) bn_elu_apply(T* __restrict__ buf, long long n, float invN,
                                                    const float* __restrict__ gsum,
                                                    const float* __restrict__ gsq,
                                                    const float* __restrict__ gamma,
                                                    const float* __restrict__ beta) {
  long long g = (long long)blockIdx.x * blockDim.x + threadIdx.x;
  long long stride = (long long)gridDim.x * blockDim.x;
  for (long long idx = g; idx < n; idx += stride) {
    int c = (int)(idx & 63);
    float mu = gsum[c] * invN;
    float var = gsq[c] * invN - mu * mu;
    float rs = rsqrtf(var + BN_EPS);
    float sc = gamma[c] * rs;
    float bi = beta[c] - mu * sc;
    float v = (float)buf[idx] * sc + bi;
    v = v > 0.f ? v : (expf(v) - 1.f);
    buf[idx] = (T)v;
  }
}

// ---------------- 27-tap gather conv: out = sum_k Y[neigh[:,k]] @ Wc[k] ----
// Each wave: 2 M-tiles (32 children) x 64 channels; B fragments shared by the
// two tiles -> 16 independent WMMAs per tap; next tap's gathered rows
// prefetched to overlap L2 latency with the WMMA burst.
__global__ void __launch_bounds__(256) conv27(const bf16_t* __restrict__ Ybf,
                                              const int* __restrict__ neigh,
                                              const bf16_t* __restrict__ Bp,
                                              float* __restrict__ out) {
  int lane = threadIdx.x & 31;
  int w = threadIdx.x >> 5;
  int l15 = lane & 15, h = lane >> 4;
  long long base = ((long long)blockIdx.x * 8 + w) * 32;
  const int* nrow0 = neigh + (base + l15) * 27;        // tile-0 row of this lane
  const int* nrow1 = neigh + (base + 16 + l15) * 27;   // tile-1 row of this lane

  v8f acc[8];                                          // [mt*4 + nt]
#pragma unroll
  for (int a = 0; a < 8; ++a)
#pragma unroll
    for (int j = 0; j < 8; ++j) acc[a][j] = 0.f;

  for (int t = 0; t < 27; ++t) {
    const bf16_t* yr0 = Ybf + (long long)nrow0[t] * 64;
    const bf16_t* yr1 = Ybf + (long long)nrow1[t] * 64;
    if (t + 1 < 27) {                                  // hide gather latency
      const bf16_t* p0 = Ybf + (long long)nrow0[t + 1] * 64;
      const bf16_t* p1 = Ybf + (long long)nrow1[t + 1] * 64;
      __builtin_prefetch(p0, 0, 3);  __builtin_prefetch(p0 + 32, 0, 3);
      __builtin_prefetch(p1, 0, 3);  __builtin_prefetch(p1 + 32, 0, 3);
    }
#pragma unroll
    for (int kc = 0; kc < 2; ++kc) {                   // K = kc*32
      Frag16 a0, a1;
      const uint4* aq0 = (const uint4*)(yr0 + kc * 32 + 8 * h);
      a0.q[0] = aq0[0];                                // K = kc*32 + 8h .. +7
      a0.q[1] = aq0[2];                                // K = kc*32 + 16 + 8h ..
      const uint4* aq1 = (const uint4*)(yr1 + kc * 32 + 8 * h);
      a1.q[0] = aq1[0];
      a1.q[1] = aq1[2];
#pragma unroll
      for (int nt = 0; nt < 4; ++nt) {
        const uint4* bq =
            (const uint4*)(Bp + ((size_t)(((t * 4 + nt) * 2 + kc)) * 32 + lane) * 16);
        Frag16 b; b.q[0] = bq[0]; b.q[1] = bq[1];
        acc[nt]     = wmma_bf16(a0, b, acc[nt]);
        acc[4 + nt] = wmma_bf16(a1, b, acc[4 + nt]);
      }
    }
  }
#pragma unroll
  for (int mt = 0; mt < 2; ++mt)
#pragma unroll
    for (int nt = 0; nt < 4; ++nt)
#pragma unroll
      for (int j = 0; j < 8; ++j) {
        long long m = base + mt * 16 + j + 8 * h;
        out[m * 64 + nt * 16 + l15] = acc[mt * 4 + nt][j];
      }
}

extern "C" void kernel_launch(void* const* d_in, const int* in_sizes, int n_in,
                              void* d_out, int out_size, void* d_ws, size_t ws_size,
                              hipStream_t stream) {
  (void)n_in; (void)out_size; (void)ws_size;
  const float* x      = (const float*)d_in[0];
  const int*   neigh  = (const int*)d_in[1];
  const float* Wup    = (const float*)d_in[2];
  const float* Wconv  = (const float*)d_in[3];
  const float* gamma1 = (const float*)d_in[4];
  const float* beta1  = (const float*)d_in[5];
  const float* gamma2 = (const float*)d_in[6];
  const float* beta2  = (const float*)d_in[7];
  float* out = (float*)d_out;

  const long long N_parent = (long long)in_sizes[0] / C_IN;   // 65536
  const long long N_child  = 8 * N_parent;                    // 524288
  const long long NC       = N_child * C_OUT;

  char* ws = (char*)d_ws;
  bf16_t* Ybf = (bf16_t*)ws;                                  // 67 MB, L2-resident
  size_t off = (size_t)NC * sizeof(bf16_t);
  float* stats = (float*)(ws + off);                          // sum1,sq1,sum2,sq2
  off += 1024;
  bf16_t* BupP = (bf16_t*)(ws + off);                         // 128 KB
  off += (size_t)32 * 4 * 32 * 16 * sizeof(bf16_t);
  bf16_t* BconvP = (bf16_t*)(ws + off);                       // 216 KB

  float* sum1 = stats, *sq1 = stats + 64, *sum2 = stats + 128, *sq2 = stats + 192;
  const float invN = 1.f / (float)N_child;

  hipMemsetAsync(stats, 0, 1024, stream);
  pack_wup  <<<16, 256, 0, stream>>>(Wup, BupP);
  pack_wconv<<<27, 256, 0, stream>>>(Wconv, BconvP);

  gemm_up<<<(int)(N_parent / 32), 256, 0, stream>>>(x, BupP, Ybf);
  bn_stats<bf16_t><<<128, 256, 0, stream>>>(Ybf, N_child, sum1, sq1);
  bn_elu_apply<bf16_t><<<2048, 256, 0, stream>>>(Ybf, NC, invN, sum1, sq1, gamma1, beta1);

  conv27<<<(int)(N_child / 256), 256, 0, stream>>>(Ybf, neigh, BconvP, out);
  bn_stats<float><<<128, 256, 0, stream>>>(out, N_child, sum2, sq2);
  bn_elu_apply<float><<<2048, 256, 0, stream>>>(out, NC, invN, sum2, sq2, gamma2, beta2);
}
